// BatchMultigraphAttention_21947282882932
// MI455X (gfx1250) — matrix operand order repaired and checked by reading, not compiled, not run on hardware
//
#include <hip/hip_runtime.h>
#include <hip/hip_bf16.h>

#define BB 8
#define NN 1024
#define FF 64
#define HH 4
#define EE 4
#define FC 32              // F_
#define ALPHA 0.2f
#define LOG2E 1.4426950408889634f

typedef __attribute__((ext_vector_type(16))) _Float16 v16h;
typedef __attribute__((ext_vector_type(8)))  float    v8f;

// ---------------------------------------------------------------------------
// Phase 1: feats = X@W per (b,h); write f16 transposed copy featsT[b,h,c,n]
// plus log2e-prescaled logit vectors attS2[b,h,n], attN2[b,h,e,n].
// One wave per (b,h,n): lane = output channel c (0..31).
// ---------------------------------------------------------------------------
__global__ __launch_bounds__(256) void gat_proj_kernel(
    const float* __restrict__ X, const float* __restrict__ W,
    const float* __restrict__ a_self, const float* __restrict__ a_neigh,
    _Float16* __restrict__ featsT, float* __restrict__ attS2,
    float* __restrict__ attN2)
{
    const int lane = threadIdx.x & 31;
    const int wv   = threadIdx.x >> 5;        // 0..7 : n within tile
    const int blk  = blockIdx.x;              // (b*H + h)*128 + ntile
    const int n0   = (blk & 127) * 8;
    const int bh   = blk >> 7;
    const int h    = bh & (HH - 1);
    const int n    = n0 + wv;
    const int c    = lane;
    const int b    = bh / HH;

    const float* xr = X + ((size_t)(b * NN + n)) * FF;
    const float* wc = W + (size_t)h * FF * FC + c;

    float acc = 0.0f;
#pragma unroll
    for (int f = 0; f < FF; ++f)
        acc = fmaf(xr[f], wc[f * FC], acc);

    // f16 transposed V for WMMA B-operand: featsT[(bh*FC + c)*N + n]
    featsT[((size_t)bh * FC + c) * NN + n] = (_Float16)acc;

    // logit dot-products (reduce over c within the wave), pre-scaled by log2e
    float vs = acc * a_self[h * FC + c];
    float v0 = acc * a_neigh[((h * EE) + 0) * FC + c];
    float v1 = acc * a_neigh[((h * EE) + 1) * FC + c];
    float v2 = acc * a_neigh[((h * EE) + 2) * FC + c];
    float v3 = acc * a_neigh[((h * EE) + 3) * FC + c];
#pragma unroll
    for (int off = 16; off > 0; off >>= 1) {
        vs += __shfl_xor(vs, off, 32);
        v0 += __shfl_xor(v0, off, 32);
        v1 += __shfl_xor(v1, off, 32);
        v2 += __shfl_xor(v2, off, 32);
        v3 += __shfl_xor(v3, off, 32);
    }
    if (lane == 0) {
        attS2[(size_t)bh * NN + n] = vs * LOG2E;
        attN2[((size_t)bh * EE + 0) * NN + n] = v0 * LOG2E;
        attN2[((size_t)bh * EE + 1) * NN + n] = v1 * LOG2E;
        attN2[((size_t)bh * EE + 2) * NN + n] = v2 * LOG2E;
        attN2[((size_t)bh * EE + 3) * NN + n] = v3 * LOG2E;
    }
}

// ---------------------------------------------------------------------------
// Phase 1b: Tmax[b,h,e] = max_n attN2[b,h,e,n].  One wave per (b,h,e).
// ---------------------------------------------------------------------------
__global__ __launch_bounds__(32) void gat_tmax_kernel(
    const float* __restrict__ attN2, float* __restrict__ Tmax)
{
    const int lane = threadIdx.x;
    const float* p = attN2 + (size_t)blockIdx.x * NN;
    float mx = -3.0e38f;
#pragma unroll
    for (int k = 0; k < NN / 32; ++k)
        mx = fmaxf(mx, p[k * 32 + lane]);
#pragma unroll
    for (int off = 16; off > 0; off >>= 1)
        mx = fmaxf(mx, __shfl_xor(mx, off, 32));
    if (lane == 0) Tmax[blockIdx.x] = mx;
}

// ---------------------------------------------------------------------------
// Phase 2: fused masked-softmax attention, single pass (per-row shift
// M_i = lrelu(s_i + max_j t_j) is a provable upper bound; softmax is
// shift-invariant, so no online rescale is needed).  Block = 4 waves =
// 4 heads on one (b, e, 16-row i-tile); each wave builds 16x32 P tiles in
// the WMMA f16 A-operand layout.  j unrolled x2 -> 4 WMMAs per iteration.
// ---------------------------------------------------------------------------
__global__ __launch_bounds__(128) void gat_attn_kernel(
    const float* __restrict__ A, const float* __restrict__ bias,
    const _Float16* __restrict__ featsT, const float* __restrict__ attS2,
    const float* __restrict__ attN2, const float* __restrict__ Tmax,
    float* __restrict__ out)
{
    const int lane = threadIdx.x & 31;
    const int h    = threadIdx.x >> 5;            // wave id == head
    const int blk  = blockIdx.x;                  // b*(E*64) + e*64 + itile
    const int it   = blk & 63;
    const int e    = (blk >> 6) & (EE - 1);
    const int b    = blk >> 8;
    const int i0   = it * 16;

    const int row   = lane & 15;                  // matrix row handled by lane
    const int hi    = lane >> 4;                  // lane-pair half (0/1)
    const int koff  = hi << 3;                    // A-operand K offset: 0 or 8
    const int joffB = hi << 4;                    // unused (B operand also uses koff)

    const float*    Arow = A     + (((size_t)(b * EE + e)) * NN + (i0 + row)) * NN;
    const float*    tn   = attN2 + ((size_t)((b * HH + h) * EE + e)) * NN;
    const _Float16* V0   = featsT + ((size_t)(b * HH + h) * FC + row)      * NN;
    const _Float16* V1   = featsT + ((size_t)(b * HH + h) * FC + 16 + row) * NN;

    // per-row shift: M = lrelu(s + T)  (log2-domain; lrelu commutes with *L)
    const float s2 = attS2[(size_t)(b * HH + h) * NN + i0 + row];
    const float T2 = Tmax[(b * HH + h) * EE + e];
    const float zc = s2 + T2;
    const float M2 = fmaxf(zc, ALPHA * zc);
    const float u1 = s2 - M2;                     // d = max(t+u1, 0.2t+u2)
    const float u2 = fmaf(ALPHA, s2, -M2);

    v8f acc0 = {}, acc1 = {};
    float rs0 = 0.0f, rs1 = 0.0f, rs2 = 0.0f, rs3 = 0.0f;  // 4 partial sums

    for (int j0 = 0; j0 < NN; j0 += 64) {
#pragma unroll
        for (int u = 0; u < 2; ++u) {             // two 32-j sub-steps
            const int js = j0 + u * 32;
            // ---- mask + neighbor logits for this lane's 16 j values ------
            const float* ap = Arow + js + koff;
            const float* tp = tn   + js + koff;
            float av[16], tv[16];
            ((float4*)av)[0] = *(const float4*)(ap +  0);
            ((float4*)av)[1] = *(const float4*)(ap +  4);
            ((float4*)av)[2] = *(const float4*)(ap + 16);
            ((float4*)av)[3] = *(const float4*)(ap + 20);
            ((float4*)tv)[0] = *(const float4*)(tp +  0);
            ((float4*)tv)[1] = *(const float4*)(tp +  4);
            ((float4*)tv)[2] = *(const float4*)(tp + 16);
            ((float4*)tv)[3] = *(const float4*)(tp + 20);

            // speculative prefetch of upcoming A tile (global_prefetch_b8)
            __builtin_prefetch((const void*)(Arow + js + 128), 0, 1);

            // ---- p = mask * 2^(lrelu(s+t) - M) ---------------------------
            float p[16];
#pragma unroll
            for (int k = 0; k < 16; ++k) {
                const float d  = fmaxf(tv[k] + u1, fmaf(ALPHA, tv[k], u2));
                const float ex = __builtin_amdgcn_exp2f(d);   // v_exp_f32
                p[k] = av[k] * ex;                            // exact 0 masked
            }
            // 4 independent accumulation chains (no serial VALU_DEP stalls)
#pragma unroll
            for (int k = 0; k < 4; ++k) {
                rs0 += p[k];
                rs1 += p[4 + k];
                rs2 += p[8 + k];
                rs3 += p[12 + k];
            }

            // ---- pack P into WMMA f16 A-operand layout -------------------
            v16h pa;
#pragma unroll
            for (int q = 0; q < 8; ++q) {
                pa[q]     = (_Float16)p[q];       // K = koff + q
                pa[8 + q] = (_Float16)p[8 + q];   // K = koff + 16 + q
            }

            // ---- B operand: two 16B runs (K = koff+q, koff+16+q) from transposed f16 feats -----
            v16h b0, b1;
#pragma unroll
            for (int q = 0; q < 8; ++q) {
                b0[q]     = V0[js + koff + q];
                b0[8 + q] = V0[js + koff + 16 + q];
                b1[q]     = V1[js + koff + q];
                b1[8 + q] = V1[js + koff + 16 + q];
            }

            acc0 = __builtin_amdgcn_wmma_f32_16x16x32_f16(
                       false, pa, false, b0, (short)0, acc0, false, false);
            acc1 = __builtin_amdgcn_wmma_f32_16x16x32_f16(
                       false, pa, false, b1, (short)0, acc1, false, false);
        }
    }

    // ---- softmax denominator: combine lane pair, broadcast per C-row -----
    const float rs = (rs0 + rs1) + (rs2 + rs3);
    const float l  = rs + __shfl_xor(rs, 16, 32);

#pragma unroll
    for (int r = 0; r < 8; ++r) {
        const float li  = __shfl(l, r + (hi << 3), 32);   // row r / r+8 sum
        const float inv = 1.0f / li;
        const int   M   = r + (hi << 3);
        const int   i   = i0 + M;
        float* op = out + (((size_t)b * NN + i) * HH + h) * (EE * FC) + e * FC;
        const float b0z = bias[h * (EE * FC) + e * FC + row];
        const float b1z = bias[h * (EE * FC) + e * FC + 16 + row];
        float o0 = acc0[r] * inv + b0z;
        float o1 = acc1[r] * inv + b1z;
        op[row]      = o0 > 0.0f ? o0 : 0.0f;
        op[16 + row] = o1 > 0.0f ? o1 : 0.0f;
    }
}

// ---------------------------------------------------------------------------
extern "C" void kernel_launch(void* const* d_in, const int* in_sizes, int n_in,
                              void* d_out, int out_size, void* d_ws, size_t ws_size,
                              hipStream_t stream) {
    const float* X       = (const float*)d_in[0];
    const float* A       = (const float*)d_in[1];
    const float* W       = (const float*)d_in[2];
    const float* a_self  = (const float*)d_in[3];
    const float* a_neigh = (const float*)d_in[4];
    const float* bias    = (const float*)d_in[5];
    float*       out     = (float*)d_out;

    // workspace carve-up
    _Float16* featsT = (_Float16*)d_ws;                                   // 2 MB
    float*    attS2  = (float*)((char*)d_ws + (size_t)BB*HH*FC*NN*2);     // 128 KB
    float*    attN2  = attS2 + (size_t)BB*HH*NN;                          // 512 KB
    float*    TmaxP  = attN2 + (size_t)BB*HH*EE*NN;                       // 512 B

    // Phase 1: projection + prescaled logits.  grid = B*H*(N/8), 8 waves/blk
    gat_proj_kernel<<<BB * HH * (NN / 8), 256, 0, stream>>>(
        X, W, a_self, a_neigh, featsT, attS2, attN2);

    // Phase 1b: per-(b,h,e) neighbor-logit max (row-shift upper bound)
    gat_tmax_kernel<<<BB * HH * EE, 32, 0, stream>>>(attN2, TmaxP);

    // Phase 2: fused masked-softmax attention + WMMA aggregation.
    // grid = B*E*(N/16), block = 128 (4 waves = 4 heads)
    gat_attn_kernel<<<BB * EE * (NN / 16), 128, 0, stream>>>(
        A, bias, featsT, attS2, attN2, TmaxP, out);
}
